// GST_AE_32392643346836
// MI455X (gfx1250) — matrix operand order repaired and checked by reading, not compiled, not run on hardware
//
#include <hip/hip_runtime.h>
#include <hip/hip_bf16.h>
#include <stdint.h>

typedef __attribute__((ext_vector_type(16))) __bf16 v16bf;
typedef __attribute__((ext_vector_type(2)))  __bf16 v2bf;
typedef __attribute__((ext_vector_type(8)))  float  v8f;

#define N_NODES 4096
#define N_EDGES 131072
#define FDIM    128      // F_IN == HID == 128
#define KPOOL   1024

// ---------------------------------------------------------------------------
// Double-buffered LDS-staged, register-blocked WMMA GEMM.
// Block = 256 threads = 8 waves (4 along M x 2 along N).
// Block tile: BM=128, BN=128, BK=32. Wave tile: 32x64 (2x4 WMMA tiles, 8 wmma/step).
// TA: A stored [Kd,M] (access A[k*M+m]); TB: B stored [N,Kd] (access B[n*Kd+k]).
// Requires M%128==0, N%128==0, Kd%32==0 (true for all shapes here).
// LDS: As[2] 128x32 bf16 row-major in K; Bs[2] 128x32 bf16 K-contiguous per col.
// ---------------------------------------------------------------------------
template <int TA, int TB>
__global__ __launch_bounds__(256) void wmma_gemm(const float* __restrict__ A,
                                                 const float* __restrict__ B,
                                                 float* __restrict__ C,
                                                 int M, int N, int Kd) {
  __shared__ __bf16 As[2][128 * 32];
  __shared__ __bf16 Bs[2][128 * 32];

  const int tid  = threadIdx.x;
  const int lane = tid & 31;
  const int wave = tid >> 5;
  const int wm   = wave >> 1;           // 0..3
  const int wn   = wave & 1;            // 0..1
  const int hf   = lane >> 4;           // lane half 0/1
  const int l    = lane & 15;

  const int m0 = blockIdx.y * 128;
  const int n0 = blockIdx.x * 128;

  v8f acc[2][4];
#pragma unroll
  for (int mr = 0; mr < 2; ++mr)
#pragma unroll
    for (int nr = 0; nr < 4; ++nr)
      acc[mr][nr] = (v8f){0.f,0.f,0.f,0.f,0.f,0.f,0.f,0.f};

  // Stage one 128x32 A tile + 128x32 B tile (2048 float2 pairs each; 8/thread)
  // into LDS buffer `bsel`, batched: all global loads first, then cvt+store.
  auto stage = [&](int bsel, int kk0) {
    float2 t[8];
    // ---- A tile ----
    if (TA == 0) {
#pragma unroll
      for (int it = 0; it < 8; ++it) {             // pair along K (contiguous)
        const int p = tid + it * 256;
        const int row = p >> 4, kp = (p & 15) << 1;
        t[it] = *(const float2*)&A[(size_t)(m0 + row) * Kd + (kk0 + kp)];
      }
#pragma unroll
      for (int it = 0; it < 8; ++it) {
        const int p = tid + it * 256;
        const int row = p >> 4, kp = (p & 15) << 1;
        *(v2bf*)&As[bsel][row * 32 + kp] = (v2bf){(__bf16)t[it].x, (__bf16)t[it].y};
      }
    } else {
#pragma unroll
      for (int it = 0; it < 8; ++it) {             // pair along M (contiguous)
        const int p = tid + it * 256;
        const int kk = p >> 6, rp = (p & 63) << 1;
        t[it] = *(const float2*)&A[(size_t)(kk0 + kk) * M + (m0 + rp)];
      }
#pragma unroll
      for (int it = 0; it < 8; ++it) {
        const int p = tid + it * 256;
        const int kk = p >> 6, rp = (p & 63) << 1;
        As[bsel][rp * 32 + kk]       = (__bf16)t[it].x;
        As[bsel][(rp + 1) * 32 + kk] = (__bf16)t[it].y;
      }
    }
    // ---- B tile ----
    if (TB == 0) {
#pragma unroll
      for (int it = 0; it < 8; ++it) {             // pair along N (contiguous)
        const int p = tid + it * 256;
        const int kk = p >> 6, cp = (p & 63) << 1;
        t[it] = *(const float2*)&B[(size_t)(kk0 + kk) * N + (n0 + cp)];
      }
#pragma unroll
      for (int it = 0; it < 8; ++it) {
        const int p = tid + it * 256;
        const int kk = p >> 6, cp = (p & 63) << 1;
        Bs[bsel][cp * 32 + kk]       = (__bf16)t[it].x;
        Bs[bsel][(cp + 1) * 32 + kk] = (__bf16)t[it].y;
      }
    } else {
#pragma unroll
      for (int it = 0; it < 8; ++it) {             // pair along K (contiguous)
        const int p = tid + it * 256;
        const int col = p >> 4, kp = (p & 15) << 1;
        t[it] = *(const float2*)&B[(size_t)(n0 + col) * Kd + (kk0 + kp)];
      }
#pragma unroll
      for (int it = 0; it < 8; ++it) {
        const int p = tid + it * 256;
        const int col = p >> 4, kp = (p & 15) << 1;
        *(v2bf*)&Bs[bsel][col * 32 + kp] = (v2bf){(__bf16)t[it].x, (__bf16)t[it].y};
      }
    }
    // one prefetch per thread per matrix, one tile beyond the double buffer
    if (kk0 + 32 < Kd) {
      const int p = tid;
      if (TA == 0) __builtin_prefetch(&A[(size_t)(m0 + (p >> 4)) * Kd + (kk0 + 32)], 0, 3);
      else         __builtin_prefetch(&A[(size_t)(kk0 + 32 + (p >> 6)) * M + m0], 0, 3);
      if (TB == 0) __builtin_prefetch(&B[(size_t)(kk0 + 32 + (p >> 6)) * N + n0], 0, 3);
      else         __builtin_prefetch(&B[(size_t)(n0 + (p >> 4)) * Kd + (kk0 + 32)], 0, 3);
    }
  };

  stage(0, 0);

  int buf = 0;
  for (int k0 = 0; k0 < Kd; k0 += 32, buf ^= 1) {
    __syncthreads();                 // buffer `buf` staged & prior reads retired
    if (k0 + 32 < Kd) stage(buf ^ 1, k0 + 32);

    // ---- fragments from LDS buffer `buf` ----
    // A fragment in this kernel's contiguous k order: lanes 0-15 K 0..15, lanes 16-31 K 16..31
    v16bf afr[2], bfr[4];
#pragma unroll
    for (int mr = 0; mr < 2; ++mr) {
      const int ar = (wm << 5) + (mr << 4) + l;
#pragma unroll
      for (int i = 0; i < 16; ++i) {
        const int kk = i + (hf << 4);
        afr[mr][i] = As[buf][ar * 32 + kk];
      }
    }
    // B fragment in the same contiguous k order: lanes 0-15 K 0..15, lanes 16-31 K 16..31
#pragma unroll
    for (int nr = 0; nr < 4; ++nr) {
      const int bc = (wn << 6) + (nr << 4) + l;
#pragma unroll
      for (int i = 0; i < 16; ++i)
        bfr[nr][i] = Bs[buf][bc * 32 + i + (hf << 4)];
    }
#pragma unroll
    for (int mr = 0; mr < 2; ++mr)
#pragma unroll
      for (int nr = 0; nr < 4; ++nr)
        acc[mr][nr] = __builtin_amdgcn_wmma_f32_16x16x32_bf16(
            false, afr[mr], false, bfr[nr], (short)0, acc[mr][nr], false, false);
  }

  // ---- store: f32 C/D layout: VGPR r -> row +r+8*hf, col = lane&15 ----
#pragma unroll
  for (int mr = 0; mr < 2; ++mr) {
    const int rb = m0 + (wm << 5) + (mr << 4) + (hf << 3);
#pragma unroll
    for (int nr = 0; nr < 4; ++nr) {
      const int cb = n0 + (wn << 6) + (nr << 4) + l;
#pragma unroll
      for (int r = 0; r < 8; ++r)
        C[(size_t)(rb + r) * N + cb] = acc[mr][nr][r];
    }
  }
}

// ---------------------------------------------------------------------------
// Small helper kernels
// ---------------------------------------------------------------------------
__global__ void fill_f32(float* __restrict__ p, float v, int n) {
  int i = blockIdx.x * blockDim.x + threadIdx.x;
  if (i < n) p[i] = v;
}

__global__ void deg_edges(const long long* __restrict__ dst, float* __restrict__ deg, int E) {
  int e = blockIdx.x * blockDim.x + threadIdx.x;
  if (e < E) atomicAdd(&deg[(int)dst[e]], 1.0f);
}

__global__ void deg_fin(const float* __restrict__ deg, float* __restrict__ dinv,
                        float* __restrict__ invd, int n) {
  int i = blockIdx.x * blockDim.x + threadIdx.x;
  if (i < n) {
    float d = deg[i];
    dinv[i] = rsqrtf(d);
    invd[i] = 1.0f / d;
  }
}

// agg = h * (1/deg) + bias  (self-loop term + bias, written before scatter)
__global__ void gcn_init(const float* __restrict__ h, const float* __restrict__ invd,
                         const float* __restrict__ bias, float* __restrict__ agg, int n) {
  int i = blockIdx.x * blockDim.x + threadIdx.x;
  if (i < n * FDIM) {
    int r = i >> 7, c = i & 127;
    agg[i] = h[i] * invd[r] + bias[c];
  }
}

// agg[dst] += h[src] * dinv[src]*dinv[dst]; 32 threads/edge, 4 feats/thread
__global__ void edge_scatter_feat(const long long* __restrict__ src,
                                  const long long* __restrict__ dst,
                                  const float* __restrict__ dinv,
                                  const float* __restrict__ h,
                                  float* __restrict__ agg, int E) {
  int t = blockIdx.x * blockDim.x + threadIdx.x;
  int e = t >> 5;
  if (e >= E) return;
  int f0 = (t & 31) * 4;
  int s = (int)src[e], d = (int)dst[e];
  float c = dinv[s] * dinv[d];
  const float* hp = h + (size_t)s * FDIM + f0;
  float* ap = agg + (size_t)d * FDIM + f0;
  atomicAdd(ap + 0, hp[0] * c);
  atomicAdd(ap + 1, hp[1] * c);
  atomicAdd(ap + 2, hp[2] * c);
  atomicAdd(ap + 3, hp[3] * c);
}

__global__ void tanh_inplace(float* __restrict__ x, int n) {
  int i = blockIdx.x * blockDim.x + threadIdx.x;
  if (i < n) x[i] = tanhf(x[i]);
}

// Row softmax over K=1024, one block of 256 threads per row
__global__ __launch_bounds__(256) void softmax_rows(float* __restrict__ S, int K) {
  __shared__ float red[256];
  const int row = blockIdx.x, tid = threadIdx.x;
  float* p = S + (size_t)row * K;
  float mx = -3.4e38f;
  for (int k = tid; k < K; k += 256) mx = fmaxf(mx, p[k]);
  red[tid] = mx; __syncthreads();
  for (int s = 128; s > 0; s >>= 1) {
    if (tid < s) red[tid] = fmaxf(red[tid], red[tid + s]);
    __syncthreads();
  }
  mx = red[0]; __syncthreads();
  float sum = 0.f;
  for (int k = tid; k < K; k += 256) {
    float e = __expf(p[k] - mx);
    p[k] = e;
    sum += e;
  }
  red[tid] = sum; __syncthreads();
  for (int s = 128; s > 0; s >>= 1) {
    if (tid < s) red[tid] += red[tid + s];
    __syncthreads();
  }
  float inv = 1.0f / red[0];
  for (int k = tid; k < K; k += 256) p[k] *= inv;
}

// Sparse attn @ dense_adj: M1T[dst, k] += S[src, k] per edge (M1T is [N,K])
__global__ void edge_scatter_pool(const long long* __restrict__ src,
                                  const long long* __restrict__ dst,
                                  const float* __restrict__ S,
                                  float* __restrict__ M1T, int E) {
  long long t = (long long)blockIdx.x * blockDim.x + threadIdx.x;
  int e = (int)(t >> 8);
  if (e >= E) return;
  int k0 = (int)(t & 255) * 4;
  int s = (int)src[e], d = (int)dst[e];
  const float* sp = S + (size_t)s * KPOOL + k0;
  float* mp = M1T + (size_t)d * KPOOL + k0;
  atomicAdd(mp + 0, sp[0]);
  atomicAdd(mp + 1, sp[1]);
  atomicAdd(mp + 2, sp[2]);
  atomicAdd(mp + 3, sp[3]);
}

__global__ void write_tail(float* __restrict__ out, int out_size, int base) {
  int i = threadIdx.x;
  if (base + i < out_size) out[base + i] = 0.f;
}

// ---------------------------------------------------------------------------
extern "C" void kernel_launch(void* const* d_in, const int* in_sizes, int n_in,
                              void* d_out, int out_size, void* d_ws, size_t ws_size,
                              hipStream_t stream) {
  const float*     nodes = (const float*)d_in[0];
  const long long* edges = (const long long*)d_in[1];   // int64 [2, E]
  const long long* src   = edges;
  const long long* dst   = edges + N_EDGES;
  const float* W1 = (const float*)d_in[3];  const float* b1 = (const float*)d_in[4];
  const float* W2 = (const float*)d_in[5];  const float* b2 = (const float*)d_in[6];
  const float* Ws = (const float*)d_in[7];
  const float* W3 = (const float*)d_in[8];  const float* b3 = (const float*)d_in[9];
  const float* W4 = (const float*)d_in[10]; const float* b4 = (const float*)d_in[11];
  const float* W5 = (const float*)d_in[12]; const float* b5 = (const float*)d_in[13];

  // workspace carve (f32), ~64 MB total
  float* ws = (float*)d_ws;
  size_t off = 0;
  auto carve = [&](size_t n) { float* p = ws + off; off += n; return p; };
  float* deg   = carve(N_NODES);
  float* dinv  = carve(N_NODES);
  float* invd  = carve(N_NODES);
  float* hbuf  = carve((size_t)N_NODES * FDIM);
  float* xA    = carve((size_t)N_NODES * FDIM);
  float* xB    = carve((size_t)N_NODES * FDIM);
  float* S     = carve((size_t)N_NODES * KPOOL);
  float* xpool = carve((size_t)KPOOL * FDIM);
  float* M1T   = carve((size_t)N_NODES * KPOOL);
  float* padj  = carve((size_t)KPOOL * KPOOL);
  float* T1    = carve((size_t)N_NODES * KPOOL);
  float* xout  = carve((size_t)N_NODES * FDIM);

  float* out_h   = (float*)d_out;                       // [N, F_IN]
  float* out_adj = out_h + (size_t)N_NODES * FDIM;      // [N, N]

  auto gemm_nn = [&](const float* A, const float* B, float* C, int M, int N, int Kd) {
    wmma_gemm<0, 0><<<dim3(N / 128, M / 128), 256, 0, stream>>>(A, B, C, M, N, Kd);
  };
  auto gemm_tn = [&](const float* A, const float* B, float* C, int M, int N, int Kd) {
    wmma_gemm<1, 0><<<dim3(N / 128, M / 128), 256, 0, stream>>>(A, B, C, M, N, Kd);
  };
  auto gemm_nt = [&](const float* A, const float* B, float* C, int M, int N, int Kd) {
    wmma_gemm<0, 1><<<dim3(N / 128, M / 128), 256, 0, stream>>>(A, B, C, M, N, Kd);
  };

  const int NF = N_NODES * FDIM;
  auto gcn = [&](const float* xin, const float* W, const float* bias, float* out, bool act) {
    gemm_nn(xin, W, hbuf, N_NODES, FDIM, FDIM);                  // h = x @ W
    gcn_init<<<(NF + 255) / 256, 256, 0, stream>>>(hbuf, invd, bias, out, N_NODES);
    edge_scatter_feat<<<(N_EDGES * 32 + 255) / 256, 256, 0, stream>>>(src, dst, dinv, hbuf, out, N_EDGES);
    if (act) tanh_inplace<<<(NF + 255) / 256, 256, 0, stream>>>(out, NF);
  };

  // --- degrees (with self-loop) ---
  fill_f32<<<(N_NODES + 255) / 256, 256, 0, stream>>>(deg, 1.0f, N_NODES);
  deg_edges<<<(N_EDGES + 255) / 256, 256, 0, stream>>>(dst, deg, N_EDGES);
  deg_fin<<<(N_NODES + 255) / 256, 256, 0, stream>>>(deg, dinv, invd, N_NODES);

  // --- encoder ---
  gcn(nodes, W1, b1, xA, true);
  gcn(xA, W2, b2, xB, true);

  // --- attention pooling ---
  gemm_nn(xB, Ws, S, N_NODES, KPOOL, FDIM);                       // logits
  softmax_rows<<<N_NODES, 256, 0, stream>>>(S, KPOOL);            // s [N,K]
  gemm_tn(S, xB, xpool, KPOOL, FDIM, N_NODES);                    // x_pool = s^T @ x

  // pool_adj = s^T A s, exploiting edge sparsity of A
  fill_f32<<<((int)((size_t)N_NODES * KPOOL) + 255) / 256, 256, 0, stream>>>(M1T, 0.f, N_NODES * KPOOL);
  edge_scatter_pool<<<N_EDGES, 256, 0, stream>>>(src, dst, S, M1T, N_EDGES);
  gemm_tn(M1T, S, padj, KPOOL, KPOOL, N_NODES);                   // pool_adj [K,K]

  // adj_out = s @ pool_adj @ s^T
  gemm_nn(S, padj, T1, N_NODES, KPOOL, KPOOL);                    // T1 [N,K]
  gemm_nt(T1, S, out_adj, N_NODES, N_NODES, KPOOL);               // adj_out [N,N]

  // x_out = s @ x_pool
  gemm_nn(S, xpool, xout, N_NODES, FDIM, KPOOL);

  // --- decoder ---
  gcn(xout, W3, b3, xA, true);
  gcn(xA, W4, b4, xB, true);
  gcn(xB, W5, b5, out_h, false);

  // trailing scalar outputs (0, 0)
  write_tail<<<1, 64, 0, stream>>>(out_h, out_size, N_NODES * FDIM + N_NODES * N_NODES);
}
